// Attention_24060406792508
// MI455X (gfx1250) — matrix operand order repaired and checked by reading, not compiled, not run on hardware
//
#include <hip/hip_runtime.h>
#include <hip/hip_bf16.h>

// MI455X / gfx1250: wave32, WMMA 16x16x32 bf16 (f32 accum).
// Pipeline: f32->bf16 convert, 3 projection GEMMs (Q pre-scaled by 1/sqrt(hd),
// V stored head-transposed), flash-attention with 64-wide KV blocks
// (online softmax, P transposed via per-wave LDS), output GEMM+bias.

typedef __bf16 bf16_t;
typedef __attribute__((ext_vector_type(16))) __bf16 v16bf;
typedef __attribute__((ext_vector_type(8)))  __bf16 v8bf;
typedef __attribute__((ext_vector_type(8)))  float  v8f;

#define BATCH   2
#define SEQ     2048
#define D_MODEL 1024
#define HEADS   16
#define HD      64

__device__ __forceinline__ bf16_t f2bf(float f) {
    union { float f; unsigned u; } x; x.f = f;
    unsigned r = (x.u + 0x7FFFu + ((x.u >> 16) & 1u)) >> 16;
    union { unsigned short s; bf16_t b; } y; y.s = (unsigned short)r;
    return y.b;
}

__device__ __forceinline__ v8f wmma_bf16(v16bf a, v16bf b, v8f c) {
    // (neg_a, A, neg_b, B, c_mod, C, reuse_a, reuse_b) -> v_wmma_f32_16x16x32_bf16
    return __builtin_amdgcn_wmma_f32_16x16x32_bf16(false, a, false, b, (short)0, c, false, false);
}

// ---------------------------------------------------------------- conversion
__global__ __launch_bounds__(256) void cvt_f32_bf16(const float* __restrict__ src,
                                                    bf16_t* __restrict__ dst, int n) {
    int i = (blockIdx.x * 256 + threadIdx.x) * 4;
    if (i + 3 < n) {
        float4 f = *(const float4*)(src + i);
        dst[i + 0] = f2bf(f.x); dst[i + 1] = f2bf(f.y);
        dst[i + 2] = f2bf(f.z); dst[i + 3] = f2bf(f.w);
    } else {
        for (int k = i; k < n; ++k) dst[k] = f2bf(src[k]);
    }
}

// ------------------------------------------------------- GEMM: C = A @ W^T
// A: (M, 1024) bf16 row-major.  W: (1024, 1024) bf16 row-major (rows = out features).
// mode 0: outb[row*1024+col] = bf16(acc * outScale)   (Q pre-scaled, K)
// mode 1: outb head-transposed: vt[((b*16+h)*64+hd)*2048 + tok]   (V^T per head)
// mode 2: outf[row*1024+col] = acc + bias[col]                    (final f32 output)
__global__ __launch_bounds__(256) void gemm_xwT(const bf16_t* __restrict__ A,
                                                const bf16_t* __restrict__ W,
                                                bf16_t* __restrict__ outb,
                                                float* __restrict__ outf,
                                                const float* __restrict__ bias,
                                                int M, int mode, float outScale) {
    const int lane = threadIdx.x & 31;
    const int wid  = (blockIdx.x * 256 + threadIdx.x) >> 5;
    const int TN   = D_MODEL / 32;                 // 32 n-tiles of width 32
    const int tm   = (wid / TN) * 64;
    const int tn   = (wid % TN) * 32;
    if (tm >= M) return;
    const int hi   = lane >> 4;                    // lane group (K split)
    const int l15  = lane & 15;
    const int koff = hi ? 16 : 0;

    v8f acc[4][2];
#pragma unroll
    for (int i = 0; i < 4; ++i)
#pragma unroll
        for (int j = 0; j < 2; ++j)
#pragma unroll
            for (int r = 0; r < 8; ++r) acc[i][j][r] = 0.0f;

    for (int k0 = 0; k0 < D_MODEL; k0 += 32) {
        v16bf a[4], b[2];
#pragma unroll
        for (int i = 0; i < 4; ++i) {
            const bf16_t* ap = A + (size_t)(tm + i * 16 + l15) * D_MODEL + k0;
            union { v16bf v; v8bf h[2]; } u;
            u.h[0] = *(const v8bf*)(ap + koff);        // K = k0+{0..7}  (+16 hi lanes)
            u.h[1] = *(const v8bf*)(ap + 8 + koff);    // K = k0+{8..15} (+16 hi lanes)
            a[i] = u.v;
        }
#pragma unroll
        for (int j = 0; j < 2; ++j) {
            // B fragment: lane holds 16 contiguous K of column (tn+j*16+l15)
            const bf16_t* wp = W + (size_t)(tn + j * 16 + l15) * D_MODEL + k0 + (hi ? 16 : 0);
            b[j] = *(const v16bf*)wp;
        }
#pragma unroll
        for (int i = 0; i < 4; ++i)
#pragma unroll
            for (int j = 0; j < 2; ++j)
                acc[i][j] = wmma_bf16(a[i], b[j], acc[i][j]);
    }

#pragma unroll
    for (int i = 0; i < 4; ++i)
#pragma unroll
        for (int j = 0; j < 2; ++j)
#pragma unroll
            for (int r = 0; r < 8; ++r) {
                int row = tm + i * 16 + r + hi * 8;
                int col = tn + j * 16 + l15;
                float v = acc[i][j][r];
                if (mode == 0) {
                    outb[(size_t)row * D_MODEL + col] = f2bf(v * outScale);
                } else if (mode == 1) {
                    int bidx = row >> 11, tok = row & (SEQ - 1);
                    int h = col >> 6, hd = col & (HD - 1);
                    outb[((size_t)((bidx * HEADS + h) * HD + hd)) * SEQ + tok] = f2bf(v);
                } else {
                    outf[(size_t)row * D_MODEL + col] = v + bias[col];
                }
            }
}

// ------------------------------------------------------- flash attention
// One wave = 16 q rows for one (batch, head). 8 waves/block -> 128 q rows.
// 64-wide KV blocks: 8 score WMMAs, one joint mask+online-softmax update
// (one shuffle-reduction set per row per 64 cols), P->LDS transpose,
// 8 PV WMMAs (K=64) against head-transposed V.
__global__ __launch_bounds__(256) void attn_fa(const bf16_t* __restrict__ Q,   // pre-scaled by 1/8
                                               const bf16_t* __restrict__ K,
                                               const bf16_t* __restrict__ Vt,
                                               const unsigned char* __restrict__ mask, // jnp.bool_
                                               bf16_t* __restrict__ attn) {
    __shared__ __align__(128) bf16_t ldsP[8][16][64];   // per-wave 16x64 P tile (16 KB)

    const int lane  = threadIdx.x & 31;
    const int wave  = threadIdx.x >> 5;
    const int qbase = blockIdx.x * 128 + wave * 16;
    const int head  = blockIdx.y;
    const int batch = blockIdx.z;
    const int hi    = lane >> 4;
    const int l15   = lane & 15;
    const int koff  = hi ? 16 : 0;

    // Q A-fragments for K = 0..63 (two 16x32 fragments), natural layout
    const bf16_t* qrowp = Q + ((size_t)(batch * SEQ + qbase + l15)) * D_MODEL + head * HD;
    v16bf qa[2];
#pragma unroll
    for (int s = 0; s < 2; ++s) {
        union { v16bf v; v8bf h[2]; } u;
        u.h[0] = *(const v8bf*)(qrowp + s * 32 + koff);
        u.h[1] = *(const v8bf*)(qrowp + s * 32 + 8 + koff);
        qa[s] = u.v;
    }

    float mrow[8], lrow[8];
    v8f acc[4];
#pragma unroll
    for (int r = 0; r < 8; ++r) { mrow[r] = -1e28f; lrow[r] = 0.0f; }
#pragma unroll
    for (int j = 0; j < 4; ++j)
#pragma unroll
        for (int r = 0; r < 8; ++r) acc[j][r] = 0.0f;

    const bf16_t* kbase = K + (size_t)batch * SEQ * D_MODEL + head * HD;
    const bf16_t* vtbase = Vt + ((size_t)((batch * HEADS + head) * HD)) * SEQ;
    const unsigned char* mbase = mask + (size_t)batch * SEQ * SEQ;

    for (int tb = 0; tb < SEQ; tb += 64) {
        if (tb + 64 < SEQ) {                       // pull next K/V block into cache
            __builtin_prefetch(kbase + (size_t)(tb + 64 + l15) * D_MODEL, 0, 1);
            __builtin_prefetch(kbase + (size_t)(tb + 96 + l15) * D_MODEL, 0, 1);
            __builtin_prefetch(vtbase + (size_t)(l15) * SEQ + tb + 64, 0, 1);
        }
        // --- scores: four 16-col C tiles (VGPR r = row, lane = col)
        v8f s[4];
#pragma unroll
        for (int cb = 0; cb < 4; ++cb) {
            const bf16_t* krow = kbase + (size_t)(tb + cb * 16 + l15) * D_MODEL + (hi ? 16 : 0);
            v16bf kb0 = *(const v16bf*)(krow);
            v16bf kb1 = *(const v16bf*)(krow + 32);
            v8f z;
#pragma unroll
            for (int r = 0; r < 8; ++r) z[r] = 0.0f;
            z = wmma_bf16(qa[0], kb0, z);
            s[cb] = wmma_bf16(qa[1], kb1, z);
        }
        // --- mask + one joint online-softmax update for the 64-col block
        float corr[8];
#pragma unroll
        for (int r = 0; r < 8; ++r) {
            int row = qbase + r + (hi ? 8 : 0);
            const unsigned char* mr = mbase + (size_t)row * SEQ + tb + l15;
            float sv[4];
#pragma unroll
            for (int cb = 0; cb < 4; ++cb)
                sv[cb] = mr[cb * 16] ? s[cb][r] : -1e30f;   // Q pre-scaled -> raw score ok
            float bm = fmaxf(fmaxf(sv[0], sv[1]), fmaxf(sv[2], sv[3]));
            bm = fmaxf(bm, __shfl_xor(bm, 1));
            bm = fmaxf(bm, __shfl_xor(bm, 2));
            bm = fmaxf(bm, __shfl_xor(bm, 4));
            bm = fmaxf(bm, __shfl_xor(bm, 8));
            float mnew = fmaxf(mrow[r], bm);
            float p[4], bs = 0.0f;
#pragma unroll
            for (int cb = 0; cb < 4; ++cb) { p[cb] = __expf(sv[cb] - mnew); bs += p[cb]; }
            bs += __shfl_xor(bs, 1);
            bs += __shfl_xor(bs, 2);
            bs += __shfl_xor(bs, 4);
            bs += __shfl_xor(bs, 8);
            float c = __expf(mrow[r] - mnew);
            lrow[r] = lrow[r] * c + bs;
            mrow[r] = mnew;
            corr[r] = c;
            bf16_t* lp = &ldsP[wave][r + (hi ? 8 : 0)][l15];
#pragma unroll
            for (int cb = 0; cb < 4; ++cb) lp[cb * 16] = f2bf(p[cb]);
        }
#pragma unroll
        for (int j = 0; j < 4; ++j)
#pragma unroll
            for (int r = 0; r < 8; ++r) acc[j][r] *= corr[r];

        // --- P (C layout) -> two A fragments via per-wave LDS (hw-ordered DS ops)
        v16bf pa[2];
#pragma unroll
        for (int s2 = 0; s2 < 2; ++s2) {
            union { v16bf v; v8bf h[2]; } pu;
            pu.h[0] = *(const v8bf*)(&ldsP[wave][l15][s2 * 32 + koff]);
            pu.h[1] = *(const v8bf*)(&ldsP[wave][l15][s2 * 32 + 8 + koff]);
            pa[s2] = pu.v;
        }
        // --- O += P @ V over K=64 (V^T rows contiguous: 32B contiguous B loads)
#pragma unroll
        for (int j = 0; j < 4; ++j) {
            const bf16_t* vrow = vtbase + (size_t)(j * 16 + l15) * SEQ + tb + (hi ? 16 : 0);
            v16bf vb0 = *(const v16bf*)(vrow);
            v16bf vb1 = *(const v16bf*)(vrow + 32);
            acc[j] = wmma_bf16(pa[0], vb0, acc[j]);
            acc[j] = wmma_bf16(pa[1], vb1, acc[j]);
        }
    }

    // epilogue: normalize; fully-masked rows (l==0) -> 0 (reference "wipe")
#pragma unroll
    for (int j = 0; j < 4; ++j)
#pragma unroll
        for (int r = 0; r < 8; ++r) {
            int row = qbase + r + (hi ? 8 : 0);
            int col = head * HD + j * 16 + l15;
            float l = lrow[r];
            float o = (l > 0.0f) ? acc[j][r] / l : 0.0f;
            attn[(size_t)(batch * SEQ + row) * D_MODEL + col] = f2bf(o);
        }
}

// ---------------------------------------------------------------- launcher
extern "C" void kernel_launch(void* const* d_in, const int* in_sizes, int n_in,
                              void* d_out, int out_size, void* d_ws, size_t ws_size,
                              hipStream_t stream) {
    const float* inq  = (const float*)d_in[0];
    const float* inkv = (const float*)d_in[1];
    const unsigned char* mask = (const unsigned char*)d_in[2];  // jnp.bool_ -> 1 byte
    const float* Wq = (const float*)d_in[3];
    const float* Wk = (const float*)d_in[4];
    const float* Wv = (const float*)d_in[5];
    const float* Wo = (const float*)d_in[6];
    const float* bo = (const float*)d_in[7];
    float* out = (float*)d_out;

    const int M = BATCH * SEQ;                       // 4096 rows
    size_t off = 0;
    auto bump = [&](size_t bytes) -> char* {
        char* p = (char*)d_ws + off;
        off += (bytes + 255) & ~(size_t)255;
        return p;
    };
    bf16_t* xq  = (bf16_t*)bump((size_t)M * D_MODEL * 2);
    bf16_t* xkv = (bf16_t*)bump((size_t)M * D_MODEL * 2);
    bf16_t* wq  = (bf16_t*)bump((size_t)D_MODEL * D_MODEL * 2);
    bf16_t* wk  = (bf16_t*)bump((size_t)D_MODEL * D_MODEL * 2);
    bf16_t* wv  = (bf16_t*)bump((size_t)D_MODEL * D_MODEL * 2);
    bf16_t* wo  = (bf16_t*)bump((size_t)D_MODEL * D_MODEL * 2);
    bf16_t* Qb  = (bf16_t*)bump((size_t)M * D_MODEL * 2);
    bf16_t* Kb  = (bf16_t*)bump((size_t)M * D_MODEL * 2);
    bf16_t* Vt  = (bf16_t*)bump((size_t)M * D_MODEL * 2);   // head-transposed V
    bf16_t* Ab  = (bf16_t*)bump((size_t)M * D_MODEL * 2);   // attention output
    // total ~56 MB of d_ws

    const int nIn = M * D_MODEL, nW = D_MODEL * D_MODEL;
    dim3 blk(256);
    cvt_f32_bf16<<<(nIn / 4 + 255) / 256, blk, 0, stream>>>(inq,  xq,  nIn);
    cvt_f32_bf16<<<(nIn / 4 + 255) / 256, blk, 0, stream>>>(inkv, xkv, nIn);
    cvt_f32_bf16<<<(nW  / 4 + 255) / 256, blk, 0, stream>>>(Wq, wq, nW);
    cvt_f32_bf16<<<(nW  / 4 + 255) / 256, blk, 0, stream>>>(Wk, wk, nW);
    cvt_f32_bf16<<<(nW  / 4 + 255) / 256, blk, 0, stream>>>(Wv, wv, nW);
    cvt_f32_bf16<<<(nW  / 4 + 255) / 256, blk, 0, stream>>>(Wo, wo, nW);

    const int wavesTotal = (M / 64) * (D_MODEL / 32);       // 2048 waves
    const int gemmBlocks = wavesTotal * 32 / 256;           // 256 blocks
    // Q pre-scaled by 1/sqrt(HD) = 0.125 so attention skips the per-score multiply
    gemm_xwT<<<gemmBlocks, blk, 0, stream>>>(xq,  wq, Qb, nullptr, nullptr, M, 0, 0.125f);
    gemm_xwT<<<gemmBlocks, blk, 0, stream>>>(xkv, wk, Kb, nullptr, nullptr, M, 0, 1.0f);
    gemm_xwT<<<gemmBlocks, blk, 0, stream>>>(xkv, wv, Vt, nullptr, nullptr, M, 1, 1.0f);

    attn_fa<<<dim3(SEQ / 128, HEADS, BATCH), blk, 0, stream>>>(Qb, Kb, Vt, mask, Ab);

    gemm_xwT<<<gemmBlocks, blk, 0, stream>>>(Ab, wo, nullptr, out, bo, M, 2, 1.0f);
}
